// LinearAttention_55216099557514
// MI455X (gfx1250) — matrix operand order repaired and checked by reading, not compiled, not run on hardware
//
#include <hip/hip_runtime.h>

// ---------------- types for CDNA5 WMMA (wave32) ----------------
typedef __bf16 bf16_t;
typedef __attribute__((ext_vector_type(16))) __bf16 v16bf;
typedef __attribute__((ext_vector_type(8)))  __bf16 v8bf;
typedef __attribute__((ext_vector_type(4)))  __bf16 v4bf;
typedef __attribute__((ext_vector_type(8)))  float  v8f;
typedef __attribute__((ext_vector_type(4)))  float  v4f;

// ---------------- problem constants ----------------
constexpr int kB  = 4;
constexpr int kN  = 4096;
constexpr int kC  = 1024;
constexpr int kH  = 16;
constexpr int kD  = 64;
constexpr int kBH = kB * kH;      // 64
constexpr int kM  = kB * kN;      // 16384 rows of x
constexpr int k3C = 3 * kC;       // 3072
constexpr int kSplit = 8;         // split-K for kv aggregation
constexpr int kKVElems = kBH * kD * kD;   // 262144

// ---------------- WMMA fragment loaders -----------------------
// A (16x32 bf16, row-major [M,K]): lane L -> row L%16,
//   K values [q*16,+16), q = L>>4.
__device__ __forceinline__ v16bf load_a_frag(const bf16_t* __restrict__ p) {
  v8bf lo = *(const v8bf*)(p);
  v8bf hi = *(const v8bf*)(p + 8);
  v16bf r;
#pragma unroll
  for (int i = 0; i < 8; ++i) { r[i] = lo[i]; r[8 + i] = hi[i]; }
  return r;
}

// B (32x16 bf16, stored K-contiguous per column: Btr[N][K]):
//   lane L -> col L%16, K values [(L>>4)*16,+16) -> one 32B load.
__device__ __forceinline__ v16bf load_b_frag(const bf16_t* __restrict__ p) {
  return *(const v16bf*)(p);
}

__device__ __forceinline__ v8f wmma_bf16(v16bf a, v16bf b, v8f c) {
  return __builtin_amdgcn_wmma_f32_16x16x32_bf16(false, a, false, b,
                                                 (short)0, c, false, false);
}

__device__ __forceinline__ float elu_p1(float x) {
  return x > 0.0f ? x + 1.0f : __expf(x);
}

// ---------------- 64x64-per-wave pipelined GEMM core -----------
// C[64x64] += A[64xK] * Btr[64xK]^T, K % 64 == 0.
// acc[i][j]: m-tile i, n-tile j (each 16x16 f32 fragment).
__device__ __forceinline__ void wave_gemm_64x64(
    const bf16_t* __restrict__ A, int lda,
    const bf16_t* __restrict__ Bt, int ldb,
    int K, int row0, int col0, int lane, v8f acc[4][4]) {
  const int mrow = lane & 15, qh = lane >> 4;
  const bf16_t* ap[4];
  const bf16_t* bp[4];
#pragma unroll
  for (int i = 0; i < 4; ++i)
    ap[i] = A + (size_t)(row0 + i * 16 + mrow) * lda + qh * 16;
#pragma unroll
  for (int j = 0; j < 4; ++j)
    bp[j] = Bt + (size_t)(col0 + j * 16 + mrow) * ldb + qh * 16;

  v16bf a0[4], b0[4], a1[4], b1[4];
#pragma unroll
  for (int i = 0; i < 4; ++i) a0[i] = load_a_frag(ap[i]);
#pragma unroll
  for (int j = 0; j < 4; ++j) b0[j] = load_b_frag(bp[j]);

  for (int k0 = 0; k0 < K; k0 += 64) {
    __builtin_prefetch(ap[0] + k0 + 256, 0, 1);
    // stage k0+32 while computing k0
#pragma unroll
    for (int i = 0; i < 4; ++i) a1[i] = load_a_frag(ap[i] + k0 + 32);
#pragma unroll
    for (int j = 0; j < 4; ++j) b1[j] = load_b_frag(bp[j] + k0 + 32);
#pragma unroll
    for (int i = 0; i < 4; ++i)
#pragma unroll
      for (int j = 0; j < 4; ++j)
        acc[i][j] = wmma_bf16(a0[i], b0[j], acc[i][j]);
    // stage k0+64 while computing k0+32
    if (k0 + 64 < K) {
#pragma unroll
      for (int i = 0; i < 4; ++i) a0[i] = load_a_frag(ap[i] + k0 + 64);
#pragma unroll
      for (int j = 0; j < 4; ++j) b0[j] = load_b_frag(bp[j] + k0 + 64);
    }
#pragma unroll
    for (int i = 0; i < 4; ++i)
#pragma unroll
      for (int j = 0; j < 4; ++j)
        acc[i][j] = wmma_bf16(a1[i], b1[j], acc[i][j]);
  }
}

// ---------------- fp32 -> bf16 conversion (x4 vectorized) ------
__global__ __launch_bounds__(256) void cvt_bf16x4_kernel(const float* __restrict__ src,
                                                         bf16_t* __restrict__ dst, int n4) {
  int i = blockIdx.x * 256 + threadIdx.x;
  if (i < n4) {
    v4f f = ((const v4f*)src)[i];
    v4bf o;
#pragma unroll
    for (int j = 0; j < 4; ++j) o[j] = (bf16_t)f[j];
    ((v4bf*)dst)[i] = o;
  }
}

// ---------------- GEMM 1: qkv = x @ w_qkv^T --------------------
// Wave computes 64 rows x 64 cols. Epilogue: elu+1 on q,k; scatter to
// Q[bh][n][d], KT[bh][d][n], VT[bh][e][n] (bf16).
__global__ __launch_bounds__(256) void qkv_gemm_kernel(
    const bf16_t* __restrict__ X, const bf16_t* __restrict__ Wqkv,
    bf16_t* __restrict__ Q, bf16_t* __restrict__ KT, bf16_t* __restrict__ VT) {
  const int lane = threadIdx.x & 31;
  const int wid  = threadIdx.x >> 5;
  const int g    = blockIdx.x * 8 + wid;          // wave id
  const int mtile = g / (k3C / 64);               // 48 column groups
  const int cgrp  = g % (k3C / 64);
  const int row0 = mtile * 64;
  const int col0 = cgrp * 64;
  const int mrow = lane & 15, qh = lane >> 4;

  v8f acc[4][4] = {};
  wave_gemm_64x64(X, kC, Wqkv, kC, kC, row0, col0, lane, acc);

  // strip is 64-column aligned -> which/head uniform per wave
  const int b_idx = row0 >> 12;                   // 4096 rows per batch
  const int which = col0 >> 10;                   // 0=q, 1=k, 2=v
  const int h     = (col0 & 1023) >> 6;
  const int bh    = b_idx * kH + h;

  if (which == 0) {
#pragma unroll
    for (int i = 0; i < 4; ++i)
#pragma unroll
      for (int t = 0; t < 4; ++t) {
        int d = t * 16 + mrow;
#pragma unroll
        for (int r = 0; r < 8; ++r) {
          int n = (row0 + i * 16 + r + 8 * qh) & (kN - 1);
          Q[((size_t)bh * kN + n) * kD + d] = (bf16_t)elu_p1(acc[i][t][r]);
        }
      }
  } else if (which == 1) {
#pragma unroll
    for (int i = 0; i < 4; ++i)
#pragma unroll
      for (int t = 0; t < 4; ++t) {
        int d = t * 16 + mrow;
#pragma unroll
        for (int r = 0; r < 8; ++r) {
          int n = (row0 + i * 16 + r + 8 * qh) & (kN - 1);
          KT[((size_t)bh * kD + d) * kN + n] = (bf16_t)elu_p1(acc[i][t][r]);
        }
      }
  } else {
#pragma unroll
    for (int i = 0; i < 4; ++i)
#pragma unroll
      for (int t = 0; t < 4; ++t) {
        int e = t * 16 + mrow;
#pragma unroll
        for (int r = 0; r < 8; ++r) {
          int n = (row0 + i * 16 + r + 8 * qh) & (kN - 1);
          VT[((size_t)bh * kD + e) * kN + n] = (bf16_t)acc[i][t][r];
        }
      }
  }
}

// ---------------- GEMM 2: kv partials (split-K x8) -------------
// One wave per (bh, ksplit): full 64x64 tile over a 512-deep K chunk.
// Output f32 partials KVpart[s][bh][e][d]; deterministic (no atomics).
__global__ __launch_bounds__(32) void kv_gemm_kernel(
    const bf16_t* __restrict__ KT, const bf16_t* __restrict__ VT,
    float* __restrict__ KVpart) {
  const int bh = blockIdx.x & (kBH - 1);
  const int s  = blockIdx.x >> 6;
  const int lane = threadIdx.x & 31;
  const int mrow = lane & 15, qh = lane >> 4;
  const int kchunk = kN / kSplit;                 // 512

  const bf16_t* A  = KT + (size_t)bh * kD * kN + s * kchunk;
  const bf16_t* Bt = VT + (size_t)bh * kD * kN + s * kchunk;

  v8f acc[4][4] = {};
  wave_gemm_64x64(A, kN, Bt, kN, kchunk, 0, 0, lane, acc);

  float* out = KVpart + (size_t)s * kKVElems + (size_t)bh * kD * kD;  // [e][d]
#pragma unroll
  for (int t = 0; t < 4; ++t) {
    int e = t * 16 + mrow;
#pragma unroll
    for (int i = 0; i < 4; ++i)
#pragma unroll
      for (int r = 0; r < 8; ++r) {
        int d = i * 16 + r + 8 * qh;
        out[e * kD + d] = acc[i][t][r];
      }
  }
}

// ---------------- kv split-K reduction -> bf16 -----------------
__global__ __launch_bounds__(256) void kv_reduce_kernel(const float* __restrict__ P,
                                                        bf16_t* __restrict__ KVt) {
  int i = blockIdx.x * 256 + threadIdx.x;         // 262144 elems
  float s = 0.0f;
#pragma unroll
  for (int j = 0; j < kSplit; ++j) s += P[(size_t)j * kKVElems + i];
  KVt[i] = (bf16_t)s;
}

// ---------------- ksum[bh][d] = sum_n k ------------------------
__global__ __launch_bounds__(256) void ksum_kernel(const bf16_t* __restrict__ KT,
                                                   float* __restrict__ Ksum) {
  __shared__ float sbuf[256];
  const int bh = blockIdx.x;
  const int tid = threadIdx.x;
  const int d = tid >> 2, part = tid & 3;
  const bf16_t* p = KT + ((size_t)bh * kD + d) * kN + part * 1024;
  float sum = 0.0f;
  for (int i = 0; i < 1024; ++i) sum += (float)p[i];
  sbuf[tid] = sum;
  __syncthreads();
  if (part == 0)
    Ksum[bh * kD + d] = sbuf[tid] + sbuf[tid + 1] + sbuf[tid + 2] + sbuf[tid + 3];
}

// ---------------- z[bh][n] = 1/(q . ksum + 1e-6) ---------------
__global__ __launch_bounds__(256) void z_kernel(const bf16_t* __restrict__ Q,
                                                const float* __restrict__ Ksum,
                                                float* __restrict__ Z) {
  __shared__ float ks[kD];
  const int bh = blockIdx.y;
  const int n  = blockIdx.x * 256 + threadIdx.x;
  if (threadIdx.x < kD) ks[threadIdx.x] = Ksum[bh * kD + threadIdx.x];
  __syncthreads();
  const bf16_t* qrow = Q + ((size_t)bh * kN + n) * kD;
  float s = 0.0f;
#pragma unroll 4
  for (int d = 0; d < kD; ++d) s += (float)qrow[d] * ks[d];
  Z[(size_t)bh * kN + n] = 1.0f / (s + 1e-6f);
}

// ---------------- GEMM 3: num = q @ kv, scale by z -------------
__global__ __launch_bounds__(128) void num_gemm_kernel(
    const bf16_t* __restrict__ Q, const bf16_t* __restrict__ KVt,
    const float* __restrict__ Z, bf16_t* __restrict__ Attn) {
  const int bh = blockIdx.y;
  const int b_idx = bh >> 4, h = bh & 15;
  const int wid = threadIdx.x >> 5;
  const int lane = threadIdx.x & 31;
  const int mrow = lane & 15, qh = lane >> 4;
  const int row0 = (blockIdx.x * 4 + wid) * 64;

  const bf16_t* A  = Q + (size_t)bh * kN * kD;
  const bf16_t* Bt = KVt + (size_t)bh * kD * kD;

  v8f acc[4][4] = {};
  wave_gemm_64x64(A, kD, Bt, kD, kD, row0, 0, lane, acc);

#pragma unroll
  for (int i = 0; i < 4; ++i)
#pragma unroll
    for (int r = 0; r < 8; ++r) {
      int n = row0 + i * 16 + r + 8 * qh;
      float zv = Z[(size_t)bh * kN + n];
      bf16_t* orow = Attn + ((size_t)(b_idx * kN + n)) * kC + h * kD;
#pragma unroll
      for (int t = 0; t < 4; ++t) orow[t * 16 + mrow] = (bf16_t)(acc[i][t][r] * zv);
    }
}

// ---------------- GEMM 4: out = attn @ w_proj^T + b ------------
__global__ __launch_bounds__(256) void proj_gemm_kernel(
    const bf16_t* __restrict__ Attn, const bf16_t* __restrict__ Wp,
    const float* __restrict__ Bias, float* __restrict__ Out) {
  const int lane = threadIdx.x & 31;
  const int wid  = threadIdx.x >> 5;
  const int g = blockIdx.x * 8 + wid;
  const int mtile = g >> 4;                  // 16 column groups (1024/64)
  const int cgrp  = g & 15;
  const int row0 = mtile * 64;
  const int col0 = cgrp * 64;
  const int mrow = lane & 15, qh = lane >> 4;

  v8f acc[4][4] = {};
  wave_gemm_64x64(Attn, kC, Wp, kC, kC, row0, col0, lane, acc);

#pragma unroll
  for (int t = 0; t < 4; ++t) {
    int col = col0 + t * 16 + mrow;
    float bv = Bias[col];
#pragma unroll
    for (int i = 0; i < 4; ++i)
#pragma unroll
      for (int r = 0; r < 8; ++r) {
        int m = row0 + i * 16 + r + 8 * qh;
        Out[(size_t)m * kC + col] = acc[i][t][r] + bv;
      }
  }
}

// ---------------- host launcher --------------------------------
extern "C" void kernel_launch(void* const* d_in, const int* in_sizes, int n_in,
                              void* d_out, int out_size, void* d_ws, size_t ws_size,
                              hipStream_t stream) {
  (void)in_sizes; (void)n_in; (void)out_size; (void)ws_size;

  const float* x      = (const float*)d_in[0];   // [4,4096,1024]
  const float* w_qkv  = (const float*)d_in[1];   // [3072,1024]
  const float* w_proj = (const float*)d_in[2];   // [1024,1024]
  const float* b_proj = (const float*)d_in[3];   // [1024]
  float* out = (float*)d_out;

  // workspace partition (256B aligned slices)
  char* w = (char*)d_ws;
  auto alloc = [&](size_t bytes) -> void* {
    void* p = (void*)w;
    w += (bytes + 255) & ~(size_t)255;
    return p;
  };
  bf16_t* Xb     = (bf16_t*)alloc((size_t)kM * kC * 2);        // 33.5 MB
  bf16_t* Wqb    = (bf16_t*)alloc((size_t)k3C * kC * 2);       //  6.3 MB
  bf16_t* Wpb    = (bf16_t*)alloc((size_t)kC * kC * 2);        //  2.1 MB
  bf16_t* Q      = (bf16_t*)alloc((size_t)kBH * kN * kD * 2);  // 33.5 MB
  bf16_t* KT     = (bf16_t*)alloc((size_t)kBH * kD * kN * 2);  // 33.5 MB
  bf16_t* VT     = (bf16_t*)alloc((size_t)kBH * kD * kN * 2);  // 33.5 MB
  float*  KVpart = (float*)alloc((size_t)kSplit * kKVElems * 4); // 8.4 MB
  bf16_t* KVt    = (bf16_t*)alloc((size_t)kKVElems * 2);       //  0.5 MB
  float*  Ksum   = (float*)alloc((size_t)kBH * kD * 4);
  float*  Z      = (float*)alloc((size_t)kBH * kN * 4);
  bf16_t* Attn   = Xb;   // alias: x_bf16 dead after qkv GEMM

  // 1) fp32 -> bf16 conversions (vectorized x4)
  cvt_bf16x4_kernel<<<(kM * kC / 4) / 256, 256, 0, stream>>>(x, Xb, kM * kC / 4);
  cvt_bf16x4_kernel<<<(k3C * kC / 4) / 256, 256, 0, stream>>>(w_qkv, Wqb, k3C * kC / 4);
  cvt_bf16x4_kernel<<<(kC * kC / 4) / 256, 256, 0, stream>>>(w_proj, Wpb, kC * kC / 4);

  // 2) qkv projection + elu+1 + transpose scatter
  //    waves = (16384/64) * (3072/64) = 12288, 8 waves/block
  qkv_gemm_kernel<<<12288 / 8, 256, 0, stream>>>(Xb, Wqb, Q, KT, VT);

  // 3) kv aggregation: split-K x8, one wave per (bh, s)
  kv_gemm_kernel<<<kBH * kSplit, 32, 0, stream>>>(KT, VT, KVpart);
  kv_reduce_kernel<<<kKVElems / 256, 256, 0, stream>>>(KVpart, KVt);

  // 4) k row-sums, 5) normalizers
  ksum_kernel<<<kBH, 256, 0, stream>>>(KT, Ksum);
  z_kernel<<<dim3(kN / 256, kBH), 256, 0, stream>>>(Q, Ksum, Z);

  // 6) numerator GEMM + normalize -> attn [B,N,C] bf16
  num_gemm_kernel<<<dim3(kN / 256, kBH), 128, 0, stream>>>(Q, KVt, Z, Attn);

  // 7) output projection + bias (fp32 out)
  //    waves = (16384/64) * (1024/64) = 4096, 8 waves/block
  proj_gemm_kernel<<<4096 / 8, 256, 0, stream>>>(Attn, Wpb, b_proj, out);
}
